// LowRank_14809047236695
// MI455X (gfx1250) — compile-verified, matrix operand order checked by reading, not run on hardware
//
#include <hip/hip_runtime.h>
#include <hip/hip_bf16.h>

typedef __attribute__((ext_vector_type(16))) __bf16 v16bf;
typedef __attribute__((ext_vector_type(8)))  float  v8f;

#define B_DIM 64
#define M_DIM 1024
#define D_DIM 1024
#define H_DIM 8
#define HD_DIM 128
#define CELU_ALPHA 1.3f
#define GN_EPS 1e-5f

static __device__ __forceinline__ unsigned short f2bf(float f) {
    unsigned int u = __float_as_uint(f);
    unsigned int r = 0x7FFFu + ((u >> 16) & 1u);
    return (unsigned short)((u + r) >> 16);
}
static __device__ __forceinline__ float bf2f(unsigned int bits16) {
    return __uint_as_float(bits16 << 16);
}

// ---------------------------------------------------------------------------
// Fused projection: Y = GroupNorm(CELU(X @ W + b)) * g + beta
// Block tile: 64 rows x 256 cols (two head groups) -> A re-read per 2 heads.
// 8 waves (2x4), each computes a 32x64 subtile via 2x4 WMMA 16x16x32 bf16.
// GroupNorm stats computed from accumulators via half-wave shuffles (C/D
// layout: lanes 0-15 / 16-31 hold different rows), combined across the two
// 64-col wave halves of each head through small LDS partials.
// ---------------------------------------------------------------------------
template<bool BF16OUT>
__global__ __launch_bounds__(256) void proj_kernel(
    const float* __restrict__ X, const float* __restrict__ W,
    const float* __restrict__ bias, const float* __restrict__ g,
    const float* __restrict__ beta, void* __restrict__ Yv)
{
    __shared__ unsigned short lds_a[64][34];    // [row][k]   bf16
    __shared__ unsigned short lds_b[256][34];   // [n][k]     bf16 (transposed)
    __shared__ float ps[64][4], ps2[64][4];     // per-row partials, one per wc
    __shared__ float smu[64][2], srs[64][2];    // per (row, head-in-tile)

    const int t    = threadIdx.x;
    const int lane = t & 31;
    const int w    = t >> 5;
    const int wr   = w >> 2;          // 0..1 : row 32-block
    const int wc   = w & 3;           // 0..3 : col 64-block
    const int r0   = blockIdx.x * 64;
    const int col0 = blockIdx.y * 256;
    const int hsel  = lane >> 4;      // 0 or 1
    const int kbase = hsel * 8;
    const int l16   = lane & 15;

    v8f acc[2][4];
#pragma unroll
    for (int i = 0; i < 2; ++i)
#pragma unroll
        for (int j = 0; j < 4; ++j) acc[i][j] = (v8f){0,0,0,0,0,0,0,0};

    const int arow = t >> 2;          // A-stage: row 0..63
    const int akq  = (t & 3) * 8;     // k offset 0,8,16,24
    const int bkr  = t >> 3;          // B-stage: k row 0..31
    const int bcq  = (t & 7) * 32;    // col chunk (32 cols per thread)

    for (int kk = 0; kk < D_DIM; kk += 32) {
        // stage A tile 64x32 (f32 -> bf16)
        {
            const float* src = X + (size_t)(r0 + arow) * D_DIM + kk + akq;
            float4 f0 = *(const float4*)(src);
            float4 f1 = *(const float4*)(src + 4);
            unsigned short* dst = &lds_a[arow][akq];
            dst[0] = f2bf(f0.x); dst[1] = f2bf(f0.y);
            dst[2] = f2bf(f0.z); dst[3] = f2bf(f0.w);
            dst[4] = f2bf(f1.x); dst[5] = f2bf(f1.y);
            dst[6] = f2bf(f1.z); dst[7] = f2bf(f1.w);
        }
        // stage B tile 32x256 transposed (f32 -> bf16)
        {
            const float* src = W + (size_t)(kk + bkr) * D_DIM + col0 + bcq;
#pragma unroll
            for (int i = 0; i < 32; i += 4) {
                float4 f = *(const float4*)(src + i);
                lds_b[bcq + i + 0][bkr] = f2bf(f.x);
                lds_b[bcq + i + 1][bkr] = f2bf(f.y);
                lds_b[bcq + i + 2][bkr] = f2bf(f.z);
                lds_b[bcq + i + 3][bkr] = f2bf(f.w);
            }
        }
        __syncthreads();

        union { v16bf v; unsigned int u[8]; } afr[2], bfr[4];
#pragma unroll
        for (int i = 0; i < 2; ++i) {
            const int m = wr * 32 + i * 16 + l16;
#pragma unroll
            for (int e = 0; e < 8; ++e) {
                const int k0 = kbase + (e & 3) * 2 + (e >> 2) * 16;
                afr[i].u[e] = *(const unsigned int*)&lds_a[m][k0];
            }
        }
#pragma unroll
        for (int j = 0; j < 4; ++j) {
            const int n = wc * 64 + j * 16 + l16;
#pragma unroll
            for (int e = 0; e < 8; ++e) {
                const int k0 = kbase + (e & 3) * 2 + (e >> 2) * 16;
                bfr[j].u[e] = *(const unsigned int*)&lds_b[n][k0];
            }
        }
#pragma unroll
        for (int i = 0; i < 2; ++i)
#pragma unroll
            for (int j = 0; j < 4; ++j)
                acc[i][j] = __builtin_amdgcn_wmma_f32_16x16x32_bf16(
                    false, afr[i].v, false, bfr[j].v, (short)0, acc[i][j],
                    false, false);
        __syncthreads();
    }

    // bias + CELU in registers
#pragma unroll
    for (int j = 0; j < 4; ++j) {
        const float bcol = bias[col0 + wc * 64 + j * 16 + l16];
#pragma unroll
        for (int i = 0; i < 2; ++i) {
#pragma unroll
            for (int v = 0; v < 8; ++v) {
                float x = acc[i][j][v] + bcol;
                acc[i][j][v] = x > 0.f
                    ? x
                    : CELU_ALPHA * (__expf(x * (1.0f / CELU_ALPHA)) - 1.0f);
            }
        }
    }

    // per-row partial sums over this wave's 64 cols (reduce across 16 lanes
    // of each half-wave; lanes 0-15 and 16-31 hold different rows)
#pragma unroll
    for (int i = 0; i < 2; ++i) {
#pragma unroll
        for (int v = 0; v < 8; ++v) {
            float s  = acc[i][0][v] + acc[i][1][v] + acc[i][2][v] + acc[i][3][v];
            float s2 = acc[i][0][v] * acc[i][0][v] + acc[i][1][v] * acc[i][1][v]
                     + acc[i][2][v] * acc[i][2][v] + acc[i][3][v] * acc[i][3][v];
#pragma unroll
            for (int o = 1; o < 16; o <<= 1) {
                s  += __shfl_xor(s,  o, 32);
                s2 += __shfl_xor(s2, o, 32);
            }
            if (l16 == 0) {
                const int lr = wr * 32 + i * 16 + v + hsel * 8;
                ps[lr][wc]  = s;
                ps2[lr][wc] = s2;
            }
        }
    }
    __syncthreads();

    // combine the two 64-col halves of each head group -> mean / rstd
    if (t < 128) {
        const int r = t >> 1, hh = t & 1;
        float s  = ps[r][2 * hh]  + ps[r][2 * hh + 1];
        float s2 = ps2[r][2 * hh] + ps2[r][2 * hh + 1];
        float mu = s * (1.f / 128.f);
        float var = s2 * (1.f / 128.f) - mu * mu;
        smu[r][hh] = mu;
        srs[r][hh] = rsqrtf(var + GN_EPS);
    }
    __syncthreads();

    // normalize + affine, store straight from registers
    const int hh = wc >> 1;
#pragma unroll
    for (int j = 0; j < 4; ++j) {
        const int col = col0 + wc * 64 + j * 16 + l16;
        const float gc = g[col], bc = beta[col];
#pragma unroll
        for (int i = 0; i < 2; ++i) {
#pragma unroll
            for (int v = 0; v < 8; ++v) {
                const int lr = wr * 32 + i * 16 + v + hsel * 8;
                float y = (acc[i][j][v] - smu[lr][hh]) * srs[lr][hh] * gc + bc;
                const size_t idx = (size_t)(r0 + lr) * D_DIM + col;
                if (BF16OUT) ((unsigned short*)Yv)[idx] = f2bf(y);
                else         ((float*)Yv)[idx] = y;
            }
        }
    }
}

// ---------------------------------------------------------------------------
// Attention tail, one block per (b,h).
// Rank-1 trick: score@W1 = cos[m] * (q_dir@W1), so am[m,j]=relu(cos*t_j+b1_j).
// ---------------------------------------------------------------------------
static __device__ __forceinline__ float block_sum(float v, float* red) {
    const int t = threadIdx.x;
    red[t] = v; __syncthreads();
    for (int s = 128; s > 0; s >>= 1) {
        if (t < s) red[t] += red[t + s];
        __syncthreads();
    }
    float r = red[0]; __syncthreads();
    return r;
}
static __device__ __forceinline__ float block_max(float v, float* red) {
    const int t = threadIdx.x;
    red[t] = v; __syncthreads();
    for (int s = 128; s > 0; s >>= 1) {
        if (t < s) red[t] = fmaxf(red[t], red[t + s]);
        __syncthreads();
    }
    float r = red[0]; __syncthreads();
    return r;
}

__global__ __launch_bounds__(256) void attn_kernel(
    const float* __restrict__ qbuf, const float* __restrict__ v1buf,
    const unsigned short* __restrict__ kbuf,
    const unsigned short* __restrict__ v2buf,
    const float* __restrict__ mask,
    const float* __restrict__ W1, const float* __restrict__ b1,
    const float* __restrict__ Wl, const float* __restrict__ bl,
    const float* __restrict__ Wl2, const float* __restrict__ bl2,
    float* __restrict__ out)
{
    __shared__ float qv[128], cosv[1024], p[1024], smask[1024];
    __shared__ float tj[64], sb1[64], sWl[64];
    __shared__ float poolp[4][64], poolv[64], v2p[2][128];
    __shared__ float red[256];
    __shared__ float sQn;

    const int t = threadIdx.x, lane = t & 31, w = t >> 5;
    const int b = blockIdx.x >> 3, h = blockIdx.x & 7;
    const size_t rowbase = (size_t)b * D_DIM + h * HD_DIM;
    const size_t kmbase  = (size_t)b * M_DIM * D_DIM + h * HD_DIM;

    if (t < 128) qv[t] = qbuf[rowbase + t];
#pragma unroll
    for (int s = 0; s < 4; ++s)
        smask[t + 256 * s] = mask[(size_t)b * M_DIM + t + 256 * s];
    if (t < 64) { sb1[t] = b1[t]; sWl[t] = Wl[t]; }
    __syncthreads();

    // ||q||
    if (t < 32) {
        float v = 0.f;
#pragma unroll
        for (int i = 0; i < 4; ++i) { float x = qv[lane + 32 * i]; v += x * x; }
        for (int o = 16; o; o >>= 1) v += __shfl_xor(v, o, 32);
        if (lane == 0) sQn = sqrtf(v);
    }
    __syncthreads();
    const float qn = sQn;

    // t_j = q_dir @ W1   (128x64 matvec, threads 0..63)
    if (t < 64) {
        const float inv = 1.0f / fmaxf(qn, 1e-12f);
        float acc = 0.f;
        for (int d = 0; d < 128; ++d) acc += qv[d] * inv * W1[d * 64 + t];
        tj[t] = acc;
    }

    // cos[m] = (q . k_m) / (max(qn,1e-8) * max(||k_m||,1e-8)); one wave per 128 m
    {
        const float q0 = qv[lane * 4 + 0], q1 = qv[lane * 4 + 1];
        const float q2 = qv[lane * 4 + 2], q3 = qv[lane * 4 + 3];
        for (int i = 0; i < 128; ++i) {
            const int m = w * 128 + i;
            const unsigned short* kp = kbuf + kmbase + (size_t)m * D_DIM + lane * 4;
            uint2 raw = *(const uint2*)kp;
            float k0 = bf2f(raw.x & 0xffffu), k1 = bf2f(raw.x >> 16);
            float k2 = bf2f(raw.y & 0xffffu), k3 = bf2f(raw.y >> 16);
            float dd  = q0 * k0 + q1 * k1 + q2 * k2 + q3 * k3;
            float kn2 = k0 * k0 + k1 * k1 + k2 * k2 + k3 * k3;
            for (int o = 16; o; o >>= 1) {
                dd  += __shfl_xor(dd, o, 32);
                kn2 += __shfl_xor(kn2, o, 32);
            }
            if (lane == 0) {
                float knv = sqrtf(kn2);
                cosv[m] = dd / (fmaxf(qn, 1e-8f) * fmaxf(knv, 1e-8f));
            }
        }
    }
    __syncthreads();

    // mask sum
    float mloc = 0.f;
#pragma unroll
    for (int s = 0; s < 4; ++s) mloc += smask[t + 256 * s];
    const float msum = block_sum(mloc, red);

    // logits a_sp[m] = sum_j relu(cos*t_j + b1_j)*Wl_j + bl, masked
    const float blv = bl[0];
    float localmax = -1e30f;
#pragma unroll
    for (int s = 0; s < 4; ++s) {
        const int m = t + 256 * s;
        const float c = cosv[m];
        float a = blv;
        for (int j = 0; j < 64; ++j) {
            float am = fmaxf(c * tj[j] + sb1[j], 0.f);
            a += am * sWl[j];
        }
        const float lgt = (smask[m] == 0.f) ? -1e9f : a;
        p[m] = lgt;
        localmax = fmaxf(localmax, lgt);
    }

    // pooled relu features: pool[j] = sum_m relu(cos*t_j+b1_j)*mask[m]
    {
        const int j = t & 63, seg = t >> 6;
        const float tjv = tj[j], b1v = sb1[j];
        float acc = 0.f;
        for (int i = 0; i < 256; ++i) {
            const int m = seg * 256 + i;
            float am = fmaxf(cosv[m] * tjv + b1v, 0.f);
            acc += am * smask[m];
        }
        poolp[seg][j] = acc;
    }

    // softmax over m
    const float gmax = block_max(localmax, red);   // also publishes poolp
    float lsum = 0.f;
#pragma unroll
    for (int s = 0; s < 4; ++s) {
        const int m = t + 256 * s;
        float e = __expf(p[m] - gmax);
        p[m] = e;
        lsum += e;
    }
    const float gsum = block_sum(lsum, red);

    if (t < 64)
        poolv[t] = (poolp[0][t] + poolp[1][t] + poolp[2][t] + poolp[3][t]) / msum;
    __syncthreads();

    // channel gate (threads 0..127 keep it in a register)
    float ach = 0.f;
    if (t < 128) {
        float acc = bl2[t];
        for (int j = 0; j < 64; ++j) acc += poolv[j] * Wl2[j * 128 + t];
        ach = 1.f / (1.f + __expf(-acc));
    }

    // v2o[d] = sum_m softmax_m * v2[m,d]
    {
        const int d = t & 127, half = t >> 7;
        float acc = 0.f;
        for (int i = 0; i < 512; ++i) {
            const int m = half * 512 + i;
            float v = bf2f((unsigned int)v2buf[kmbase + (size_t)m * D_DIM + d]);
            acc += p[m] * v;
        }
        v2p[half][d] = acc;
    }
    __syncthreads();
    if (t < 128) {
        float v2o = (v2p[0][t] + v2p[1][t]) / gsum;
        out[rowbase + t] = v1buf[rowbase + t] * v2o * ach;
    }
}

// ---------------------------------------------------------------------------
extern "C" void kernel_launch(void* const* d_in, const int* in_sizes, int n_in,
                              void* d_out, int out_size, void* d_ws, size_t ws_size,
                              hipStream_t stream) {
    (void)in_sizes; (void)n_in; (void)out_size; (void)ws_size;
    const float* query  = (const float*)d_in[0];
    const float* key    = (const float*)d_in[1];
    const float* mask   = (const float*)d_in[2];
    const float* value1 = (const float*)d_in[3];
    const float* value2 = (const float*)d_in[4];
    const float* W_q = (const float*)d_in[5],  *b_q = (const float*)d_in[6],
               * g_q = (const float*)d_in[7],  *be_q = (const float*)d_in[8];
    const float* W_k = (const float*)d_in[9],  *b_k = (const float*)d_in[10],
               * g_k = (const float*)d_in[11], *be_k = (const float*)d_in[12];
    const float* W_v1 = (const float*)d_in[13], *b_v1 = (const float*)d_in[14],
               * g_v1 = (const float*)d_in[15], *be_v1 = (const float*)d_in[16];
    const float* W_v2 = (const float*)d_in[17], *b_v2 = (const float*)d_in[18],
               * g_v2 = (const float*)d_in[19], *be_v2 = (const float*)d_in[20];
    const float* W1  = (const float*)d_in[21], *b1  = (const float*)d_in[22];
    const float* Wl  = (const float*)d_in[23], *bl  = (const float*)d_in[24];
    const float* Wl2 = (const float*)d_in[25], *bl2 = (const float*)d_in[26];

    char* ws = (char*)d_ws;
    float*          qbuf  = (float*)(ws);                               // 256 KB
    float*          v1buf = (float*)(ws + (size_t)262144);              // 256 KB
    unsigned short* kbuf  = (unsigned short*)(ws + (size_t)524288);     // 128 MB
    unsigned short* v2buf = (unsigned short*)(ws + (size_t)524288
                                                 + (size_t)134217728);  // 128 MB

    dim3 blk(256);
    proj_kernel<false><<<dim3(1, 4),    blk, 0, stream>>>(query,  W_q,  b_q,  g_q,  be_q,  (void*)qbuf);
    proj_kernel<false><<<dim3(1, 4),    blk, 0, stream>>>(value1, W_v1, b_v1, g_v1, be_v1, (void*)v1buf);
    proj_kernel<true ><<<dim3(1024, 4), blk, 0, stream>>>(key,    W_k,  b_k,  g_k,  be_k,  (void*)kbuf);
    proj_kernel<true ><<<dim3(1024, 4), blk, 0, stream>>>(value2, W_v2, b_v2, g_v2, be_v2, (void*)v2buf);

    attn_kernel<<<dim3(B_DIM * H_DIM), blk, 0, stream>>>(
        qbuf, v1buf, kbuf, v2buf, mask, W1, b1, Wl, bl, Wl2, bl2, (float*)d_out);
}
